// TransformerLayerTorchAttention_20117626814577
// MI455X (gfx1250) — matrix operand order repaired and checked by reading, not compiled, not run on hardware
//
#include <hip/hip_runtime.h>
#include <hip/hip_bf16.h>

// ---------------------------------------------------------------------------
// Problem constants (from reference): B=16, N=1024, D=256, H=8, FF=1024, dh=32
// ---------------------------------------------------------------------------
#define BB   16
#define NN   1024
#define DD   256
#define HH   8
#define FFD  1024
#define DH   32
#define MM   (BB * NN)          // 16384 rows
#define BH   (BB * HH)          // 128 (batch*heads)

typedef _Float16 f16;
typedef __attribute__((ext_vector_type(8)))  _Float16 v8h;
typedef __attribute__((ext_vector_type(16))) _Float16 v16h;
typedef __attribute__((ext_vector_type(8)))  float    v8f;
typedef __attribute__((ext_vector_type(4)))  int      v4i;

// CDNA5 async Global->LDS staging (ASYNCcnt) if the toolchain exposes it.
#if __has_builtin(__builtin_amdgcn_global_load_async_to_lds_b128) && \
    __has_builtin(__builtin_amdgcn_s_wait_asynccnt)
#define USE_ASYNC_LDS 1
#else
#define USE_ASYNC_LDS 0
#endif

// Builtin signature (from hipcc diagnostic): (v4i AS1*, v4i AS3*, imm, imm)
#define AS1V4(p) ((__attribute__((address_space(1))) v4i*)(p))
#define AS3V4(p) ((__attribute__((address_space(3))) v4i*)(p))

// ---------------------------------------------------------------------------
// WMMA fragment helpers (wave32, V_WMMA_F32_16X16X32_F16)
//
// A-matrix 16x32 f16 (ISA 7.12.2): lane L (row = L&15, half = L>>4):
//   elems 0..7  = K = half*8 + 0..7 ; elems 8..15 = K = 16 + half*8 + 0..7
// B-matrix 32x16 f16: lane L (col = L&15, half = L>>4):
//   elems 0..7  = K = half*8 + 0..7 ; elems 8..15 = K = 16 + half*8 + 0..7
// C/D 16x16 f32: VGPR r, lane L -> row = r + 8*(L>>4), col = L&15
// ---------------------------------------------------------------------------
__device__ __forceinline__ v16h cat8(v8h lo, v8h hi) {
  return __builtin_shufflevector(lo, hi, 0,1,2,3,4,5,6,7,8,9,10,11,12,13,14,15);
}

__device__ __forceinline__ v16h load_fragA(const f16* p, int ld) {
  const int lane = threadIdx.x & 31;
  const f16* r = p + (lane & 15) * ld + (lane >> 4) * 8;
  v8h lo = *(const v8h*)(r);
  v8h hi = *(const v8h*)(r + 16);
  return cat8(lo, hi);
}

// B fragment for x @ W^T: source rows ARE the output columns, K contiguous.
__device__ __forceinline__ v16h load_fragB(const f16* p, int ld) {
  const int lane = threadIdx.x & 31;
  const f16* r = p + (lane & 15) * ld + (lane >> 4) * 8;
  v8h lo = *(const v8h*)(r);
  v8h hi = *(const v8h*)(r + 16);
  return cat8(lo, hi);
}

__device__ __forceinline__ v8f wmma_f16(v16h a, v16h b, v8f c) {
  return __builtin_amdgcn_wmma_f32_16x16x32_f16(false, a, false, b,
                                                (short)0, c, false, false);
}

// ---------------------------------------------------------------------------
// f32 -> f16 conversion
// ---------------------------------------------------------------------------
__global__ void cvt_f32_to_f16(const float* __restrict__ x,
                               f16* __restrict__ y, int n) {
  int i = blockIdx.x * blockDim.x + threadIdx.x;
  if (i < n) y[i] = (f16)x[i];
}

// ---------------------------------------------------------------------------
// Register-blocked, LDS double-buffered GEMM core.
//   out[M,Nout] = A[M,K](f16) @ Bw[Nout,K]^T(f16)
// Block = 256 threads = 8 waves in a 2(M) x 4(N) grid.
// Block tile 64x128, wave tile 32x32 = 2x2 WMMA accumulators.
// One barrier per 32-wide K-step; next step's loads overlap the WMMAs.
// Staging uses GLOBAL_LOAD_ASYNC_TO_LDS_B128 (ASYNCcnt) when available,
// else b128 reg-staging. LDS row stride 40 halves (80 B) vs bank conflicts.
// ---------------------------------------------------------------------------
#define LDSTRIDE 40

struct GemmCtx {
  int wm, wn, m0, n0;
  int arow, aseg, brow, bseg;
};

__device__ __forceinline__ GemmCtx gemm_ctx() {
  GemmCtx c;
  const int tid = threadIdx.x;
  const int wave = tid >> 5;
  c.wm = wave >> 2;               // 0..1
  c.wn = wave & 3;                // 0..3
  c.m0 = blockIdx.x * 64;
  c.n0 = blockIdx.y * 128;
  c.arow = tid >> 2;              // 0..63   (A: 64 rows x 32 halves)
  c.aseg = (tid & 3) * 8;         // halves
  c.brow = tid >> 1;              // 0..127  (B: 128 rows x 32 halves)
  c.bseg = (tid & 1) * 16;        // halves
  return c;
}

// Runs the K-loop, leaving 2x2 f32 accumulators. Abuf/Bbuf are LDS.
__device__ __forceinline__ void gemm_mainloop(
    const GemmCtx& c, const f16* __restrict__ A, const f16* __restrict__ Bw,
    int K, f16* Abuf0, f16* Abuf1, f16* Bbuf0, f16* Bbuf1, v8f acc[2][2]) {
  f16* Ab[2] = {Abuf0, Abuf1};
  f16* Bb[2] = {Bbuf0, Bbuf1};

  const f16* agp = &A[c.arow * K + c.aseg];        // += 32 per K-step
  const f16* bgp = &Bw[c.brow * K + c.bseg];
  f16* asp = Abuf0 + c.arow * LDSTRIDE + c.aseg;   // LDS store ptrs (buf 0)
  f16* bsp = Bbuf0 + c.brow * LDSTRIDE + c.bseg;
  const size_t lbo = (size_t)(Abuf1 - Abuf0);      // A buffer offset (B buffers are 2x this apart)
  const int ksteps = K >> 5;

#if USE_ASYNC_LDS
  // ---- async cache->LDS staging, no VGPR round trip ----
  auto issue = [&](int kt, int buf) {
    const int ko = kt << 5;
    const size_t bo = buf ? lbo : 0;
    __builtin_amdgcn_global_load_async_to_lds_b128(AS1V4(agp + ko),
                                                   AS3V4(asp + bo), 0, 0);
    __builtin_amdgcn_global_load_async_to_lds_b128(AS1V4(bgp + ko),
                                                   AS3V4(bsp + bo), 0, 0);
    __builtin_amdgcn_global_load_async_to_lds_b128(AS1V4(bgp + ko + 8),
                                                   AS3V4(bsp + bo + 8), 0, 0);
  };
  issue(0, 0);
  __builtin_amdgcn_s_wait_asynccnt(0);
  __syncthreads();
  for (int kt = 0; kt < ksteps; ++kt) {
    const int cur = kt & 1;
    const bool has_next = (kt + 1) < ksteps;
    if (has_next) {
      issue(kt + 1, 1 - cur);                      // overlap with WMMAs
      if (kt + 2 < ksteps) {                       // global_prefetch_b8
        __builtin_prefetch(agp + ((kt + 2) << 5), 0, 3);
        __builtin_prefetch(bgp + ((kt + 2) << 5), 0, 3);
      }
    }
    const f16* Ac = Ab[cur];
    const f16* Bc = Bb[cur];
    v16h a0 = load_fragA(Ac + (c.wm * 32 + 0)  * LDSTRIDE, LDSTRIDE);
    v16h a1 = load_fragA(Ac + (c.wm * 32 + 16) * LDSTRIDE, LDSTRIDE);
    v16h b0 = load_fragB(Bc + (c.wn * 32 + 0)  * LDSTRIDE, LDSTRIDE);
    v16h b1 = load_fragB(Bc + (c.wn * 32 + 16) * LDSTRIDE, LDSTRIDE);
    acc[0][0] = wmma_f16(a0, b0, acc[0][0]);
    acc[0][1] = wmma_f16(a0, b1, acc[0][1]);
    acc[1][0] = wmma_f16(a1, b0, acc[1][0]);
    acc[1][1] = wmma_f16(a1, b1, acc[1][1]);
    if (has_next) {
      __builtin_amdgcn_s_wait_asynccnt(0);
      __syncthreads();
    }
  }
#else
  // ---- fallback: b128 reg staging ----
  v8h ra, rb0, rb1;
  ra  = *(const v8h*)(agp);
  rb0 = *(const v8h*)(bgp);
  rb1 = *(const v8h*)(bgp + 8);
  *(v8h*)(asp) = ra;
  *(v8h*)(bsp) = rb0;
  *(v8h*)(bsp + 8) = rb1;
  __syncthreads();

  for (int kt = 0; kt < ksteps; ++kt) {
    const int cur = kt & 1;
    const bool has_next = (kt + 1) < ksteps;
    if (has_next) {
      const int ko = (kt + 1) << 5;
      ra  = *(const v8h*)(agp + ko);
      rb0 = *(const v8h*)(bgp + ko);
      rb1 = *(const v8h*)(bgp + ko + 8);
      if (kt + 2 < ksteps) {
        __builtin_prefetch(agp + ((kt + 2) << 5), 0, 3);
        __builtin_prefetch(bgp + ((kt + 2) << 5), 0, 3);
      }
    }
    const f16* Ac = Ab[cur];
    const f16* Bc = Bb[cur];
    v16h a0 = load_fragA(Ac + (c.wm * 32 + 0)  * LDSTRIDE, LDSTRIDE);
    v16h a1 = load_fragA(Ac + (c.wm * 32 + 16) * LDSTRIDE, LDSTRIDE);
    v16h b0 = load_fragB(Bc + (c.wn * 32 + 0)  * LDSTRIDE, LDSTRIDE);
    v16h b1 = load_fragB(Bc + (c.wn * 32 + 16) * LDSTRIDE, LDSTRIDE);
    acc[0][0] = wmma_f16(a0, b0, acc[0][0]);
    acc[0][1] = wmma_f16(a0, b1, acc[0][1]);
    acc[1][0] = wmma_f16(a1, b0, acc[1][0]);
    acc[1][1] = wmma_f16(a1, b1, acc[1][1]);
    if (has_next) {
      const size_t nb = cur ? 0 : lbo;
      *(v8h*)(asp + nb) = ra;
      *(v8h*)(bsp + nb) = rb0;
      *(v8h*)(bsp + nb + 8) = rb1;
      __syncthreads();
    }
  }
#endif
}

// Generic epilogue variant: + bias, optional residual, optional ReLU,
// f32 and/or f16 outputs.
template <bool RELU, bool RES>
__global__ __launch_bounds__(256) void gemm_xwT(
    const f16* __restrict__ A, const f16* __restrict__ Bw,
    const float* __restrict__ bias, const float* __restrict__ res,
    float* __restrict__ out32, f16* __restrict__ out16,
    int M, int Nout, int K) {
  __shared__ f16 Abuf[2][64 * LDSTRIDE];
  __shared__ f16 Bbuf[2][128 * LDSTRIDE];
  const GemmCtx c = gemm_ctx();
  v8f acc[2][2] = {};
  gemm_mainloop(c, A, Bw, K, Abuf[0], Abuf[1], Bbuf[0], Bbuf[1], acc);

  const int lane = threadIdx.x & 31;
  const int colb = lane & 15;
  const int rowb = (lane >> 4) * 8;
#pragma unroll
  for (int i = 0; i < 2; ++i) {
#pragma unroll
    for (int j = 0; j < 2; ++j) {
      const int nc = c.n0 + c.wn * 32 + j * 16 + colb;
      const float bv = bias ? bias[nc] : 0.0f;
#pragma unroll
      for (int r = 0; r < 8; ++r) {
        const int mr = c.m0 + c.wm * 32 + i * 16 + rowb + r;
        float v = acc[i][j][r] + bv;
        if (RES)  v += res[mr * Nout + nc];
        if (RELU) v = v > 0.0f ? v : 0.0f;
        if (out32) out32[mr * Nout + nc] = v;
        if (out16) out16[mr * Nout + nc] = (f16)v;
      }
    }
  }
}

// QKV variant: epilogue scatters per-head Q,K -> [bh, n, 32] and V transposed
// -> [bh, 32, n] so every attention fragment is a contiguous row load.
__global__ __launch_bounds__(256) void gemm_qkv(
    const f16* __restrict__ A, const f16* __restrict__ Bw,
    const float* __restrict__ bias,
    f16* __restrict__ qb, f16* __restrict__ kb, f16* __restrict__ vtb) {
  __shared__ f16 Abuf[2][64 * LDSTRIDE];
  __shared__ f16 Bbuf[2][128 * LDSTRIDE];
  const GemmCtx c = gemm_ctx();
  v8f acc[2][2] = {};
  gemm_mainloop(c, A, Bw, DD, Abuf[0], Abuf[1], Bbuf[0], Bbuf[1], acc);

  const int lane = threadIdx.x & 31;
  const int colb = lane & 15;
  const int rowb = (lane >> 4) * 8;
#pragma unroll
  for (int i = 0; i < 2; ++i) {
#pragma unroll
    for (int j = 0; j < 2; ++j) {
      const int col = c.n0 + c.wn * 32 + j * 16 + colb;   // 0..767
      const float bv = bias[col];
#pragma unroll
      for (int r = 0; r < 8; ++r) {
        const int m = c.m0 + c.wm * 32 + i * 16 + rowb + r;
        const int b = m >> 10;
        const int idx = m & (NN - 1);
        const f16 hv = (f16)(acc[i][j][r] + bv);
        if (col < DD) {                                   // Q
          const int hh = col >> 5, d = col & 31;
          qb[(((b * HH + hh) << 10) + idx) * DH + d] = hv;
        } else if (col < 2 * DD) {                        // K
          const int cc = col - DD;
          const int hh = cc >> 5, d = cc & 31;
          kb[(((b * HH + hh) << 10) + idx) * DH + d] = hv;
        } else {                                          // V^T [bh,32,N]
          const int cc = col - 2 * DD;
          const int hh = cc >> 5, d = cc & 31;
          vtb[((b * HH + hh) * DH + d) * NN + idx] = hv;
        }
      }
    }
  }
}

// ---------------------------------------------------------------------------
// Flash-style attention: 4 independent waves per block, one (bh, 16-query)
// tile per wave, online softmax -> zero workgroup barriers, 1.25 KB LDS/wave
// (only to convert P from C-layout to A-layout between the two WMMAs).
// Per 32-key chunk: 2 score WMMAs + 2 PV WMMAs.
// ---------------------------------------------------------------------------
__global__ __launch_bounds__(128) void attn_kernel(
    const f16* __restrict__ qb, const f16* __restrict__ kb,
    const f16* __restrict__ vtb, const int* __restrict__ adj,
    f16* __restrict__ ob) {
  __shared__ f16 pstage[4][16 * LDSTRIDE];   // per-wave P-tile staging

  const int wave = threadIdx.x >> 5;
  const int lane = threadIdx.x & 31;
  const int job = blockIdx.x * 4 + wave;     // 0..8191
  const int bh = job >> 6;                   // 0..127
  const int qt = job & 63;
  const int b = bh >> 3, h = bh & 7;
  const int q0 = qt * 16;
  const int colq = lane & 15;
  const int half = lane >> 4;
  const float scale = 0.17677669529663687f;  // 1/sqrt(32)

  f16* stg = pstage[wave];
  const v16h aq = load_fragA(qb + (bh * NN + q0) * DH, DH);
  const int* adjrow = adj + (b * NN + q0) * NN;

  float run_m[8], run_l[8];
#pragma unroll
  for (int r = 0; r < 8; ++r) { run_m[r] = -3.0e38f; run_l[r] = 0.0f; }
  v8f o0 = {}, o1 = {};

  for (int k0 = 0; k0 < NN; k0 += 32) {
    // ---- scores: two 16-key tiles (K = dh = 32 in one WMMA each) ----
    v16h bk0 = load_fragB(kb + (bh * NN + k0) * DH, DH);
    v16h bk1 = load_fragB(kb + (bh * NN + k0 + 16) * DH, DH);
    v8f s0 = {}, s1 = {};
    s0 = wmma_f16(aq, bk0, s0);
    s1 = wmma_f16(aq, bk1, s1);

    // ---- mask + online softmax update (per row r+8*half) ----
    float p0[8], p1[8];
#pragma unroll
    for (int r = 0; r < 8; ++r) {
      const int row = r + 8 * half;
      float v0 = s0[r] * scale;
      float v1 = s1[r] * scale;
      if (adjrow[row * NN + k0 + colq]      != 0) v0 = -__builtin_inff();
      if (adjrow[row * NN + k0 + 16 + colq] != 0) v1 = -__builtin_inff();
      float cm = fmaxf(v0, v1);              // chunk max across 16 lanes/row
#pragma unroll
      for (int mk = 8; mk >= 1; mk >>= 1) cm = fmaxf(cm, __shfl_xor(cm, mk, 16));
      const float nm = fmaxf(run_m[r], cm);  // run_m finite => nm finite
      const float rs = __expf(run_m[r] - nm);
      p0[r] = __expf(v0 - nm);
      p1[r] = __expf(v1 - nm);
      float cs = p0[r] + p1[r];
#pragma unroll
      for (int mk = 8; mk >= 1; mk >>= 1) cs += __shfl_xor(cs, mk, 16);
      run_l[r] = run_l[r] * rs + cs;
      run_m[r] = nm;
      o0[r] *= rs;                           // rescale accumulators
      o1[r] *= rs;
    }

    // ---- C-layout -> A-layout via per-wave LDS (in-order per wave) ----
#pragma unroll
    for (int r = 0; r < 8; ++r) {
      const int row = r + 8 * half;
      stg[row * LDSTRIDE + colq]      = (f16)p0[r];
      stg[row * LDSTRIDE + 16 + colq] = (f16)p1[r];
    }
    v16h pa  = load_fragA(stg, LDSTRIDE);
    v16h bv0 = load_fragB(vtb + (bh * DH + 0)  * NN + k0, NN);
    v16h bv1 = load_fragB(vtb + (bh * DH + 16) * NN + k0, NN);
    o0 = wmma_f16(pa, bv0, o0);
    o1 = wmma_f16(pa, bv1, o1);
  }

  // ---- normalize and store [B*N, D] with head offset ----
#pragma unroll
  for (int r = 0; r < 8; ++r) {
    const int row = q0 + r + 8 * half;
    const float inv = 1.0f / run_l[r];
    const int base = (b * NN + row) * DD + h * DH;
    ob[base + colq]      = (f16)(o0[r] * inv);
    ob[base + 16 + colq] = (f16)(o1[r] * inv);
  }
}

// ---------------------------------------------------------------------------
// LayerNorm over D=256: one row per wave, wave32 __shfl_xor reductions.
// ---------------------------------------------------------------------------
__global__ __launch_bounds__(256) void ln_kernel(
    const float* __restrict__ x, const float* __restrict__ g,
    const float* __restrict__ be, float* __restrict__ out32,
    f16* __restrict__ out16) {
  const int lane = threadIdx.x & 31;
  const int row = blockIdx.x * 8 + (threadIdx.x >> 5);
  const float* xp = x + row * DD;

  float v[8];
  float s = 0.0f, s2 = 0.0f;
#pragma unroll
  for (int t = 0; t < 8; ++t) {
    float a = xp[lane + 32 * t];
    v[t] = a;
    s += a;
    s2 += a * a;
  }
#pragma unroll
  for (int m = 16; m >= 1; m >>= 1) {
    s  += __shfl_xor(s,  m, 32);
    s2 += __shfl_xor(s2, m, 32);
  }
  const float mean = s * (1.0f / DD);
  const float var  = s2 * (1.0f / DD) - mean * mean;
  const float rinv = rsqrtf(var + 1e-5f);
#pragma unroll
  for (int t = 0; t < 8; ++t) {
    const int cidx = lane + 32 * t;
    const float y = (v[t] - mean) * rinv * g[cidx] + be[cidx];
    if (out32) out32[row * DD + cidx] = y;
    if (out16) out16[row * DD + cidx] = (f16)y;
  }
}

// ---------------------------------------------------------------------------
// Launcher
// ---------------------------------------------------------------------------
extern "C" void kernel_launch(void* const* d_in, const int* in_sizes, int n_in,
                              void* d_out, int out_size, void* d_ws, size_t ws_size,
                              hipStream_t stream) {
  (void)in_sizes; (void)n_in; (void)out_size; (void)ws_size;

  const float* h     = (const float*)d_in[0];
  const int*   adj   = (const int*)  d_in[1];
  const float* w_qkv = (const float*)d_in[2];
  const float* b_qkv = (const float*)d_in[3];
  const float* w_out = (const float*)d_in[4];
  const float* b_out = (const float*)d_in[5];
  const float* w1    = (const float*)d_in[6];
  const float* b1    = (const float*)d_in[7];
  const float* w2    = (const float*)d_in[8];
  const float* b2    = (const float*)d_in[9];
  const float* g1    = (const float*)d_in[10];
  const float* be1   = (const float*)d_in[11];
  const float* g2    = (const float*)d_in[12];
  const float* be2   = (const float*)d_in[13];
  float* out = (float*)d_out;

  // ---- carve workspace ----
  size_t off = 0;
  auto carve = [&](size_t bytes) -> void* {
    off = (off + 255) & ~(size_t)255;
    void* p = (char*)d_ws + off;
    off += bytes;
    return p;
  };
  f16* h16     = (f16*)carve((size_t)MM * DD * 2);
  f16* wqkv16  = (f16*)carve((size_t)3 * DD * DD * 2);
  f16* wout16  = (f16*)carve((size_t)DD * DD * 2);
  f16* w116    = (f16*)carve((size_t)FFD * DD * 2);
  f16* w216    = (f16*)carve((size_t)DD * FFD * 2);
  f16* qb      = (f16*)carve((size_t)BH * NN * DH * 2);
  f16* kb      = (f16*)carve((size_t)BH * NN * DH * 2);
  f16* vtb     = (f16*)carve((size_t)BH * DH * NN * 2);
  f16* ob      = (f16*)carve((size_t)MM * DD * 2);
  float* y32   = (float*)carve((size_t)MM * DD * 4);
  float* h1_32 = (float*)carve((size_t)MM * DD * 4);
  f16* h1_16   = (f16*)carve((size_t)MM * DD * 2);
  f16* ff16    = (f16*)carve((size_t)MM * FFD * 2);

  // ---- 1) convert activations/weights to f16 ----
  auto cvt = [&](const float* src, f16* dst, int n) {
    cvt_f32_to_f16<<<(n + 255) / 256, 256, 0, stream>>>(src, dst, n);
  };
  cvt(h,     h16,    MM * DD);
  cvt(w_qkv, wqkv16, 3 * DD * DD);
  cvt(w_out, wout16, DD * DD);
  cvt(w1,    w116,   FFD * DD);
  cvt(w2,    w216,   DD * FFD);

  // ---- 2) QKV projection + per-head scatter (V transposed) ----
  gemm_qkv<<<dim3(MM / 64, (3 * DD) / 128), 256, 0, stream>>>(
      h16, wqkv16, b_qkv, qb, kb, vtb);

  // ---- 3) flash attention (mask + online softmax + PV) ----
  attn_kernel<<<dim3((BH * (NN / 16)) / 4), 128, 0, stream>>>(
      qb, kb, vtb, adj, ob);

  // ---- 4) output projection + residual -> y32 ----
  gemm_xwT<false, true><<<dim3(MM / 64, DD / 128), 256, 0, stream>>>(
      ob, wout16, b_out, h, y32, (f16*)nullptr, MM, DD, DD);

  // ---- 5) LayerNorm 1 -> h1 (f32 for residual, f16 for FFN input) ----
  ln_kernel<<<MM / 8, 256, 0, stream>>>(y32, g1, be1, h1_32, h1_16);

  // ---- 6) FFN up + ReLU -> ff16 ----
  gemm_xwT<true, false><<<dim3(MM / 64, FFD / 128), 256, 0, stream>>>(
      h1_16, w116, b1, (const float*)nullptr, (float*)nullptr, ff16,
      MM, FFD, DD);

  // ---- 7) FFN down + residual(h1) -> y32 ----
  gemm_xwT<false, true><<<dim3(MM / 64, DD / 128), 256, 0, stream>>>(
      ff16, w216, b2, h1_32, y32, (f16*)nullptr, MM, DD, FFD);

  // ---- 8) LayerNorm 2 -> output ----
  ln_kernel<<<MM / 8, 256, 0, stream>>>(y32, g2, be2, out, (f16*)nullptr);
}
